// imageCaptionModel_3530463117588
// MI455X (gfx1250) — matrix operand order repaired and checked by reading, not compiled, not run on hardware
//
#include <hip/hip_runtime.h>

// ---------------------------------------------------------------------------
// CDNA5 (gfx1250) wave32 WMMA bf16 implementation of the image-caption model.
// Software-pipelined fragment loads + v_wmma_f32_16x16x32_bf16 chains.
// ---------------------------------------------------------------------------

typedef __attribute__((ext_vector_type(16))) __bf16 v16bf;
typedef __attribute__((ext_vector_type(8)))  float  v8f;

#define B_  128
#define T_  30
#define V_  10000
#define E_  512
#define H_  512
#define F_  4096
#define L_  2

union Frag { uint4 u[2]; v16bf v; };

__device__ __forceinline__ unsigned short f2bf(float f) {
  unsigned int x = __float_as_uint(f);
  x += 0x7FFFu + ((x >> 16) & 1u);          // round-to-nearest-even
  return (unsigned short)(x >> 16);
}

__device__ __forceinline__ float sigmoidf_(float x) {
  return 1.0f / (1.0f + __expf(-x));
}

// Fragment loaders (this code's contiguous k order, same for A and B, wave32):
//   A 16x32: lane L holds row (L&15); 16 contiguous K values,
//            K base = (L>=16 ? 16 : 0).
//   B 32x16: lane L holds col (L&15); 16 contiguous K values,
//            K base = (L>=16 ? 16 : 0).
__device__ __forceinline__ void loadA(Frag& a, const unsigned short* aRow,
                                      int k0, int sel) {
  a.u[0] = *(const uint4*)(aRow + k0 + sel * 16);
  a.u[1] = *(const uint4*)(aRow + k0 + sel * 16 + 8);
}
__device__ __forceinline__ void loadB4(Frag b[4], const unsigned short* bRow0,
                                       int ldb, int k0) {
#pragma unroll
  for (int s = 0; s < 4; ++s) {
    const unsigned short* p = bRow0 + (size_t)(s * 16) * ldb + k0;
    b[s].u[0] = *(const uint4*)(p);
    b[s].u[1] = *(const uint4*)(p + 8);
  }
}
__device__ __forceinline__ void wmma4(const Frag& a, const Frag b[4], v8f acc[4]) {
#pragma unroll
  for (int s = 0; s < 4; ++s)
    acc[s] = __builtin_amdgcn_wmma_f32_16x16x32_bf16(
        false, a.v, false, b[s].v, (short)0, acc[s], false, false);
}

// ---------------------------------------------------------------------------
// Wave-level GEMM segment, 16(M) x 64(N) tile over K contiguous columns.
// Double-buffered: fragments for step k+1 load while step k WMMAs execute.
// Requires K % 64 == 0; issues one 32-element prefetch overshoot past the
// segment (workspace is padded).
// ---------------------------------------------------------------------------
__device__ __forceinline__ void gemm_seg(
    const unsigned short* __restrict__ A, int lda,
    const unsigned short* __restrict__ BT, int ldb,
    int K, int mBase, int nBase, v8f acc[4])
{
  const int lane = threadIdx.x & 31;
  const int rowA = lane & 15;
  const int sel  = lane >> 4;

  const unsigned short* aRow  = A + (size_t)(mBase + rowA) * lda;
  const unsigned short* bRow0 = BT + (size_t)(nBase + rowA) * ldb + sel * 16;

  Frag a0, a1, b0[4], b1[4];
  loadA(a0, aRow, 0, sel);
  loadB4(b0, bRow0, ldb, 0);

  for (int k0 = 0; k0 < K; k0 += 64) {
    loadA(a1, aRow, k0 + 32, sel);
    loadB4(b1, bRow0, ldb, k0 + 32);
    wmma4(a0, b0, acc);
    loadA(a0, aRow, k0 + 64, sel);          // overshoots on last iter (padded)
    loadB4(b0, bRow0, ldb, k0 + 64);
    wmma4(a1, b1, acc);
  }
}

// Dual-B variant (u and r gates): one A-fragment load feeds 8 WMMAs per
// 32-wide K step; all 18 b128 loads of a step issue before a single wait.
__device__ __forceinline__ void gemm_seg2(
    const unsigned short* __restrict__ A, int lda,
    const unsigned short* __restrict__ BTu,
    const unsigned short* __restrict__ BTr, int ldb,
    int K, int mBase, int nBase, v8f accU[4], v8f accR[4])
{
  const int lane = threadIdx.x & 31;
  const int rowA = lane & 15;
  const int sel  = lane >> 4;

  const unsigned short* aRow   = A + (size_t)(mBase + rowA) * lda;
  const unsigned short* bRowU0 = BTu + (size_t)(nBase + rowA) * ldb + sel * 16;
  const unsigned short* bRowR0 = BTr + (size_t)(nBase + rowA) * ldb + sel * 16;

  for (int k0 = 0; k0 < K; k0 += 32) {
    Frag a, bu_[4], br_[4];
    loadA(a, aRow, k0, sel);
    loadB4(bu_, bRowU0, ldb, k0);
    loadB4(br_, bRowR0, ldb, k0);
    wmma4(a, bu_, accU);
    wmma4(a, br_, accR);
  }
}

// ---------------------------------------------------------------------------
// Prep kernels
// ---------------------------------------------------------------------------
__global__ void k_cast_bf16(const float* __restrict__ in,
                            unsigned short* __restrict__ out, int n) {
  int i = blockIdx.x * blockDim.x + threadIdx.x;
  if (i < n) out[i] = f2bf(in[i]);
}

// in: f32 [K][N] row-major  ->  out: bf16 [N][K] row-major
__global__ void k_transpose_bf16(const float* __restrict__ in,
                                 unsigned short* __restrict__ out,
                                 int K, int N) {
  int i = blockIdx.x * blockDim.x + threadIdx.x;
  if (i < K * N) {
    int k = i / N, n = i % N;
    out[(size_t)n * K + k] = f2bf(in[i]);
  }
}

// x_emb bf16, time-major: out[t][b][e] = emb[tok[b][t]][e]
__global__ void k_embed(const float* __restrict__ emb,
                        const int* __restrict__ tok,
                        unsigned short* __restrict__ out) {
  int i = blockIdx.x * blockDim.x + threadIdx.x;
  if (i < T_ * B_ * E_) {
    int t = i / (B_ * E_);
    int rem = i - t * (B_ * E_);
    int b = rem / E_;
    int e = rem - b * E_;
    out[i] = f2bf(emb[(size_t)tok[b * T_ + t] * E_ + e]);
  }
}

// ---------------------------------------------------------------------------
// h0 = tanh(vgg @ W_in + b_in), replicated to both layers (f32 + bf16 copies)
// ---------------------------------------------------------------------------
__global__ __launch_bounds__(32)
void k_h0(const unsigned short* __restrict__ Avgg,       // [128][4096] bf16
          const unsigned short* __restrict__ WinT,       // [512][4096] bf16
          const float* __restrict__ b_in,
          float* __restrict__ hF32,                      // [2][128][512]
          unsigned short* __restrict__ hB16)             // [2][128][512]
{
  const int nBase = blockIdx.x * 64;
  const int mBase = blockIdx.y * 16;
  v8f acc[4];
#pragma unroll
  for (int s = 0; s < 4; ++s) acc[s] = v8f{};
  gemm_seg(Avgg, F_, WinT, F_, F_, mBase, nBase, acc);

  const int lane = threadIdx.x & 31;
  const int col  = lane & 15;
  const int sel  = lane >> 4;
#pragma unroll
  for (int s = 0; s < 4; ++s) {
    int n = nBase + s * 16 + col;
#pragma unroll
    for (int r = 0; r < 8; ++r) {
      int m = mBase + r + 8 * sel;
      float v = tanhf(acc[s][r] + b_in[n]);
      int idx = m * H_ + n;
      hF32[idx] = v;               hF32[B_ * H_ + idx] = v;
      unsigned short bv = f2bf(v);
      hB16[idx] = bv;              hB16[B_ * H_ + idx] = bv;
    }
  }
}

// ---------------------------------------------------------------------------
// GRU pass 1: u = sigmoid([x|h]@Wu + bu), r = sigmoid([x|h]@Wr + br),
//             rh = bf16(r * h)     (fused dual-B WMMA pipeline)
// ---------------------------------------------------------------------------
__global__ __launch_bounds__(32)
void k_gates(const unsigned short* __restrict__ Ax,     // [128][512] bf16
             const unsigned short* __restrict__ Ah,     // [128][512] bf16
             const unsigned short* __restrict__ WuT,    // [512][1024] bf16
             const unsigned short* __restrict__ WrT,    // [512][1024] bf16
             const float* __restrict__ bu,
             const float* __restrict__ br,
             const float* __restrict__ hF32,            // [128][512]
             float* __restrict__ uF32,                  // [128][512]
             unsigned short* __restrict__ rhB16)        // [128][512]
{
  const int nBase = blockIdx.x * 64;
  const int mBase = blockIdx.y * 16;
  v8f accU[4], accR[4];
#pragma unroll
  for (int s = 0; s < 4; ++s) { accU[s] = v8f{}; accR[s] = v8f{}; }

  // k in [0,512): x part ; k in [512,1024): h part (BT column offset 512)
  gemm_seg2(Ax, E_, WuT,       WrT,       E_ + H_, E_, mBase, nBase, accU, accR);
  gemm_seg2(Ah, H_, WuT + E_,  WrT + E_,  E_ + H_, H_, mBase, nBase, accU, accR);

  const int lane = threadIdx.x & 31;
  const int col  = lane & 15;
  const int sel  = lane >> 4;
#pragma unroll
  for (int s = 0; s < 4; ++s) {
    int n = nBase + s * 16 + col;
#pragma unroll
    for (int r = 0; r < 8; ++r) {
      int m = mBase + r + 8 * sel;
      int idx = m * H_ + n;
      float u = sigmoidf_(accU[s][r] + bu[n]);
      float rr = sigmoidf_(accR[s][r] + br[n]);
      uF32[idx] = u;
      rhB16[idx] = f2bf(rr * hF32[idx]);
    }
  }
}

// ---------------------------------------------------------------------------
// GRU pass 2: cand = tanh([x|r*h]@Wc + bc); h = u*h + (1-u)*cand
// ---------------------------------------------------------------------------
__global__ __launch_bounds__(32)
void k_cand(const unsigned short* __restrict__ Ax,      // [128][512] bf16
            const unsigned short* __restrict__ Arh,     // [128][512] bf16
            const unsigned short* __restrict__ WcT,     // [512][1024] bf16
            const float* __restrict__ bc,
            const float* __restrict__ uF32,
            float* __restrict__ hF32,                   // in/out [128][512]
            unsigned short* __restrict__ hB16,          // out [128][512]
            unsigned short* __restrict__ hist)          // optional [128][512]
{
  const int nBase = blockIdx.x * 64;
  const int mBase = blockIdx.y * 16;
  v8f acc[4];
#pragma unroll
  for (int s = 0; s < 4; ++s) acc[s] = v8f{};
  gemm_seg(Ax,  E_, WcT,      E_ + H_, E_, mBase, nBase, acc);
  gemm_seg(Arh, H_, WcT + E_, E_ + H_, H_, mBase, nBase, acc);

  const int lane = threadIdx.x & 31;
  const int col  = lane & 15;
  const int sel  = lane >> 4;
#pragma unroll
  for (int s = 0; s < 4; ++s) {
    int n = nBase + s * 16 + col;
#pragma unroll
    for (int r = 0; r < 8; ++r) {
      int m = mBase + r + 8 * sel;
      int idx = m * H_ + n;
      float cand = tanhf(acc[s][r] + bc[n]);
      float u = uF32[idx];
      float hnew = u * hF32[idx] + (1.0f - u) * cand;
      hF32[idx] = hnew;
      unsigned short bv = f2bf(hnew);
      hB16[idx] = bv;
      if (hist) hist[idx] = bv;
    }
  }
}

// ---------------------------------------------------------------------------
// Logits: [T*B, 512] @ [512, 10000] + b_out, scattered to [B][T][V].
// Tail handling: nBase clamped to V_-64 (16-aligned) => the last two chunks
// overlap and redundantly compute/store identical values.  No branches.
// ---------------------------------------------------------------------------
__global__ __launch_bounds__(32)
void k_logits(const unsigned short* __restrict__ hist,  // [T*B][512] bf16
              const unsigned short* __restrict__ WoutT, // [10000][512] bf16
              const float* __restrict__ b_out,
              float* __restrict__ out)                  // [B][T][V]
{
  int nBase = blockIdx.x * 64;
  if (nBase > V_ - 64) nBase = V_ - 64;                 // 9936, 16-aligned
  const int mBase = blockIdx.y * 16;
  v8f acc[4];
#pragma unroll
  for (int s = 0; s < 4; ++s) acc[s] = v8f{};
  gemm_seg(hist, H_, WoutT, H_, H_, mBase, nBase, acc);

  const int lane = threadIdx.x & 31;
  const int col  = lane & 15;
  const int sel  = lane >> 4;
#pragma unroll
  for (int s = 0; s < 4; ++s) {
    int n = nBase + s * 16 + col;
#pragma unroll
    for (int r = 0; r < 8; ++r) {
      int gm = mBase + r + 8 * sel;     // row = t*128 + b
      int t = gm >> 7;
      int b = gm & 127;
      out[(size_t)b * T_ * V_ + (size_t)t * V_ + n] = acc[s][r] + b_out[n];
    }
  }
}

__global__ void k_final_state(const float* __restrict__ hF32,
                              float* __restrict__ out) {
  int i = blockIdx.x * blockDim.x + threadIdx.x;
  if (i < L_ * B_ * H_) out[i] = hF32[i];
}

// ---------------------------------------------------------------------------
// Host-side orchestration
// ---------------------------------------------------------------------------
extern "C" void kernel_launch(void* const* d_in, const int* in_sizes, int n_in,
                              void* d_out, int out_size, void* d_ws, size_t ws_size,
                              hipStream_t stream) {
  (void)in_sizes; (void)n_in; (void)out_size; (void)ws_size;

  const float* vgg   = (const float*)d_in[0];
  const int*   tok   = (const int*)  d_in[1];
  const float* emb   = (const float*)d_in[2];
  const float* W_in  = (const float*)d_in[3];
  const float* b_in  = (const float*)d_in[4];
  const float* Wu    = (const float*)d_in[5];
  const float* bu    = (const float*)d_in[6];
  const float* Wr    = (const float*)d_in[7];
  const float* br    = (const float*)d_in[8];
  const float* Wc    = (const float*)d_in[9];
  const float* bc    = (const float*)d_in[10];
  const float* W_out = (const float*)d_in[11];
  const float* b_out = (const float*)d_in[12];

  char* ws = (char*)d_ws;
  size_t off = 0;
  auto alloc = [&](size_t bytes) -> char* {
    char* p = ws + off;
    off = (off + bytes + 255) & ~(size_t)255;
    return p;
  };

  unsigned short* vggB  = (unsigned short*)alloc((size_t)B_ * F_ * 2);
  unsigned short* WinT  = (unsigned short*)alloc((size_t)H_ * F_ * 2);
  unsigned short* WuT   = (unsigned short*)alloc((size_t)L_ * H_ * (E_ + H_) * 2);
  unsigned short* WrT   = (unsigned short*)alloc((size_t)L_ * H_ * (E_ + H_) * 2);
  unsigned short* WcT   = (unsigned short*)alloc((size_t)L_ * H_ * (E_ + H_) * 2);
  unsigned short* WoutT = (unsigned short*)alloc((size_t)V_ * H_ * 2);
  unsigned short* xemb  = (unsigned short*)alloc((size_t)T_ * B_ * E_ * 2);
  float*          hF32  = (float*)         alloc((size_t)L_ * B_ * H_ * 4);
  unsigned short* hB16  = (unsigned short*)alloc((size_t)L_ * B_ * H_ * 2);
  float*          uF32  = (float*)         alloc((size_t)B_ * H_ * 4);
  unsigned short* rhB16 = (unsigned short*)alloc((size_t)B_ * H_ * 2);
  unsigned short* hist  = (unsigned short*)alloc((size_t)T_ * B_ * H_ * 2);
  (void)alloc(256);   // padding: keeps the pipelined prefetch overshoot mapped

  const int TPB = 256;

  // --- prep: casts / transposes / gather ---
  k_cast_bf16<<<(B_ * F_ + TPB - 1) / TPB, TPB, 0, stream>>>(vgg, vggB, B_ * F_);
  k_transpose_bf16<<<(F_ * H_ + TPB - 1) / TPB, TPB, 0, stream>>>(W_in, WinT, F_, H_);
  for (int j = 0; j < L_; ++j) {
    const size_t wOff = (size_t)j * (E_ + H_) * H_;
    const size_t tOff = (size_t)j * H_ * (E_ + H_);
    k_transpose_bf16<<<((E_ + H_) * H_ + TPB - 1) / TPB, TPB, 0, stream>>>(
        Wu + wOff, WuT + tOff, E_ + H_, H_);
    k_transpose_bf16<<<((E_ + H_) * H_ + TPB - 1) / TPB, TPB, 0, stream>>>(
        Wr + wOff, WrT + tOff, E_ + H_, H_);
    k_transpose_bf16<<<((E_ + H_) * H_ + TPB - 1) / TPB, TPB, 0, stream>>>(
        Wc + wOff, WcT + tOff, E_ + H_, H_);
  }
  k_transpose_bf16<<<(H_ * V_ + TPB - 1) / TPB, TPB, 0, stream>>>(W_out, WoutT, H_, V_);
  k_embed<<<(T_ * B_ * E_ + TPB - 1) / TPB, TPB, 0, stream>>>(emb, tok, xemb);

  // --- h0 ---
  k_h0<<<dim3(H_ / 64, B_ / 16), 32, 0, stream>>>(vggB, WinT, b_in, hF32, hB16);

  // --- recurrent loop ---
  for (int t = 0; t < T_; ++t) {
    const unsigned short* x0 = xemb + (size_t)t * B_ * E_;
    for (int j = 0; j < L_; ++j) {
      const unsigned short* xin = (j == 0) ? x0 : hB16;        // layer1 input = h of layer0
      unsigned short* hj = hB16 + (size_t)j * B_ * H_;
      float* hFj = hF32 + (size_t)j * B_ * H_;
      const unsigned short* WuTj = WuT + (size_t)j * H_ * (E_ + H_);
      const unsigned short* WrTj = WrT + (size_t)j * H_ * (E_ + H_);
      const unsigned short* WcTj = WcT + (size_t)j * H_ * (E_ + H_);

      k_gates<<<dim3(H_ / 64, B_ / 16), 32, 0, stream>>>(
          xin, hj, WuTj, WrTj, bu + j * H_, br + j * H_, hFj, uF32, rhB16);

      unsigned short* hist_t = (j == L_ - 1) ? (hist + (size_t)t * B_ * H_) : nullptr;
      k_cand<<<dim3(H_ / 64, B_ / 16), 32, 0, stream>>>(
          xin, rhB16, WcTj, bc + j * H_, uF32, hFj, hj, hist_t);
    }
  }

  // --- batched vocabulary projection for all 30 timesteps ---
  float* out = (float*)d_out;
  k_logits<<<dim3((V_ + 63) / 64, (T_ * B_) / 16), 32, 0, stream>>>(
      hist, WoutT, b_out, out);

  // --- final hidden state [L, B, H] appended after logits ---
  k_final_state<<<(L_ * B_ * H_ + TPB - 1) / TPB, TPB, 0, stream>>>(
      hF32, out + (size_t)B_ * T_ * V_);
}
